// WaypointSmoother_80350248173878
// MI455X (gfx1250) — matrix operand order repaired and checked by reading, not compiled, not run on hardware
//
#include <hip/hip_runtime.h>
#include <math.h>

typedef __attribute__((ext_vector_type(16))) _Float16 v16h;
typedef __attribute__((ext_vector_type(8)))  _Float16 v8h;
typedef __attribute__((ext_vector_type(4)))  _Float16 v4h;
typedef __attribute__((ext_vector_type(8)))  float    v8f;

constexpr int kN       = 512;   // waypoints per trajectory
constexpr int kThreads = 256;   // 8 waves (wave32)
constexpr int kWaves   = 8;
constexpr int kTilesPW = 4;     // 32 tiles of 16 points total

__device__ __forceinline__ float sigmoidf_(float x) { return 1.0f / (1.0f + expf(-x)); }

__global__ __launch_bounds__(kThreads)
void waypoint_smoother_kernel(const float* __restrict__ wp,   // (B, N, 2)
                              const float* __restrict__ sw,   // scalar
                              const float* __restrict__ W1,   // (4, 64) row-major
                              const float* __restrict__ B1,   // (64)
                              const float* __restrict__ W2,   // (64, 32)
                              const float* __restrict__ B2,   // (32)
                              const float* __restrict__ W3,   // (32, 1)
                              const float* __restrict__ B3,   // (1)
                              float* __restrict__ out)        // (B, N, 2)
{
    __shared__ float                 ptsX[kN], ptsY[kN];
    __shared__ __align__(16) _Float16 hL[kWaves][16 * 64];  // layer-1 acts (f16, m-major)
    __shared__ float                 h2L[kWaves][16 * 32];  // layer-2 acts (f32)
    __shared__ __align__(8)  _Float16 w1t[64 * 4];          // W1 transposed: [n][k]
    __shared__ float                 b1s[64];
    __shared__ __align__(32) _Float16 w2t[32 * 64];         // W2 transposed: [n][k]
    __shared__ float                 b2s[32];
    __shared__ float                 w3s[32];
    __shared__ float                 wArr[kN];
    __shared__ float                 aggA[kWaves], aggCx[kWaves], aggCy[kWaves];
    __shared__ float                 preA[kWaves], preCx[kWaves], preCy[kWaves];
    __shared__ float                 scal2[2];              // sigmoid(sw), b3

    const int t    = threadIdx.x;
    const int bb   = blockIdx.x;
    const int lane = t & 31;
    const int wv   = t >> 5;
    const int hh   = lane >> 4;   // half-wave (0/1)
    const int lc   = lane & 15;

    // ---- stage inputs ----
    {
        const float2* wp2 = (const float2*)(wp + (size_t)bb * kN * 2);
        float2 p0 = wp2[t];        ptsX[t]       = p0.x; ptsY[t]       = p0.y;
        float2 p1 = wp2[t + 256];  ptsX[t + 256] = p1.x; ptsY[t + 256] = p1.y;
    }
    // W1 transposed: w1t[n*4 + k] = W1[k*64 + n]
    w1t[t] = (_Float16)W1[(t & 3) * 64 + (t >> 2)];
    if (t < 64) b1s[t] = B1[t];
    // W2 transposed: w2t[n*64 + k] = W2[k*32 + n]
#pragma unroll
    for (int k8 = 0; k8 < 8; ++k8) {
        const int idx = t + k8 * 256;            // idx = k*32 + n
        w2t[(idx & 31) * 64 + (idx >> 5)] = (_Float16)W2[idx];
    }
    if (t < 32) { b2s[t] = B2[t]; w3s[t] = W3[t]; }
    if (t == 0) { scal2[0] = sigmoidf_(sw[0]); scal2[1] = B3[0]; }
    __syncthreads();

    // ---- hoisted (tile-invariant) B operands and biases ----
    v16h B1V[4];                 // layer-1 B: rows 0..3 of W1, K padded to 32
    float bias1[4];
#pragma unroll
    for (int nt = 0; nt < 4; ++nt) {
        const int n = nt * 16 + lc;
        const v4h r = *(const v4h*)&w1t[n * 4];
        v16h b;
#pragma unroll
        for (int e = 0; e < 16; ++e) b[e] = (_Float16)0.0f;
        // lanes 16-31 carry K=16..31 rows -> zero; select without branching
        b[0] = (hh == 0) ? r[0] : (_Float16)0.0f;
        b[1] = (hh == 0) ? r[1] : (_Float16)0.0f;
        b[2] = (hh == 0) ? r[2] : (_Float16)0.0f;
        b[3] = (hh == 0) ? r[3] : (_Float16)0.0f;
        B1V[nt] = b;
        bias1[nt] = b1s[n];
    }
    v16h B2V[2][2];              // layer-2 B: [nt2][kk], contiguous 16 halves each
    float bias2[2];
#pragma unroll
    for (int nt2 = 0; nt2 < 2; ++nt2) {
        const int n2 = nt2 * 16 + lc;
#pragma unroll
        for (int kk = 0; kk < 2; ++kk)
            B2V[nt2][kk] = *(const v16h*)&w2t[n2 * 64 + kk * 32 + 16 * hh];
        bias2[nt2] = b2s[n2];
    }

    // ================= MLP via WMMA (f16 in, f32 accumulate) =================
    for (int tt = 0; tt < kTilesPW; ++tt) {
        const int tile = wv * kTilesPW + tt;
        const int i    = tile * 16 + lc;         // this lane's M-row point index

        // features [d1x,d1y,d2x,d2y], branch-free
        const int ia  = (i < 510) ? i : 510;
        const int im1 = (i > 0) ? i - 1 : 0;
        const int ip1 = (i < 511) ? i + 1 : 511;
        const float d1x = ptsX[ia + 1] - ptsX[ia];
        const float d1y = ptsY[ia + 1] - ptsY[ia];
        const float msk = (i >= 1 && i <= 510) ? 1.0f : 0.0f;
        const float d2x = msk * ((ptsX[ip1] - ptsX[i]) - (ptsX[i] - ptsX[im1]));
        const float d2y = msk * ((ptsY[ip1] - ptsY[i]) - (ptsY[i] - ptsY[im1]));

        // layer-1 A: lanes 0-15 hold K=0..3 (feats), everything else zero
        v16h a1;
#pragma unroll
        for (int e = 0; e < 16; ++e) a1[e] = (_Float16)0.0f;
        a1[0] = (_Float16)((hh == 0) ? d1x : 0.0f);
        a1[1] = (_Float16)((hh == 0) ? d1y : 0.0f);
        a1[2] = (_Float16)((hh == 0) ? d2x : 0.0f);
        a1[3] = (_Float16)((hh == 0) ? d2y : 0.0f);

#pragma unroll
        for (int nt = 0; nt < 4; ++nt) {
            const int n = nt * 16 + lc;
            v8f c;
#pragma unroll
            for (int v = 0; v < 8; ++v) c[v] = bias1[nt];
            c = __builtin_amdgcn_wmma_f32_16x16x32_f16(false, a1, false, B1V[nt],
                                                       (short)0, c, false, false);
            // D: VGPR v, col = lc, row = v + 8*hh ; ReLU -> f16 LDS (m-major)
#pragma unroll
            for (int v = 0; v < 8; ++v) {
                const float r = c[v] > 0.0f ? c[v] : 0.0f;
                hL[wv][(v + 8 * hh) * 64 + n] = (_Float16)r;
            }
        }

        // layer-2 A: two contiguous 16B chunks per K-chunk (k-contiguous in hL)
        v16h a2[2];
#pragma unroll
        for (int kk = 0; kk < 2; ++kk) {
            const _Float16* base = &hL[wv][lc * 64 + kk * 32 + 16 * hh];
            const v8h lo = *(const v8h*)base;
            const v8h hi = *(const v8h*)(base + 8);
#pragma unroll
            for (int e = 0; e < 8; ++e) { a2[kk][e] = lo[e]; a2[kk][8 + e] = hi[e]; }
        }

#pragma unroll
        for (int nt2 = 0; nt2 < 2; ++nt2) {
            const int n2 = nt2 * 16 + lc;
            v8f c2;
#pragma unroll
            for (int v = 0; v < 8; ++v) c2[v] = bias2[nt2];
#pragma unroll
            for (int kk = 0; kk < 2; ++kk)
                c2 = __builtin_amdgcn_wmma_f32_16x16x32_f16(false, a2[kk], false,
                                                            B2V[nt2][kk],
                                                            (short)0, c2, false, false);
#pragma unroll
            for (int v = 0; v < 8; ++v) {
                const float r = c2[v] > 0.0f ? c2[v] : 0.0f;
                h2L[wv][(v + 8 * hh) * 32 + n2] = r;
            }
        }

        // layer-3: both half-waves sum 16 terms for point m=lc, combine via shfl
        {
            float o = 0.0f;
#pragma unroll
            for (int k = 0; k < 16; ++k) {
                const int kk = 16 * hh + k;
                o += h2L[wv][lc * 32 + kk] * w3s[kk];
            }
            o += __shfl_xor(o, 16, 32);
            o += scal2[1];
            if (hh == 0) {
                const float cw = sigmoidf_(o);
                wArr[tile * 16 + lc] = scal2[0] * (1.0f - cw);
            }
        }
    }
    __syncthreads();

    // ============== parallel scan of affine maps y = a*y_prev + c ==============
    const int i0 = 2 * t, i1 = 2 * t + 1;
    float a0, c0x, c0y;
    if (i0 == 0) { a0 = 0.0f; c0x = ptsX[0]; c0y = ptsY[0]; }   // f0 = const x0
    else { a0 = wArr[i0]; c0x = (1.0f - a0) * ptsX[i0]; c0y = (1.0f - a0) * ptsY[i0]; }
    const float a1s = wArr[i1];
    const float c1x = (1.0f - a1s) * ptsX[i1];
    const float c1y = (1.0f - a1s) * ptsY[i1];

    // local composite f_{i1} o f_{i0}
    float sa  = a1s * a0;
    float scx = a1s * c0x + c1x;
    float scy = a1s * c0y + c1y;

    // wave32 inclusive scan (composition) via lane shuffles
#pragma unroll
    for (int d = 1; d < 32; d <<= 1) {
        const float pa  = __shfl_up(sa,  d, 32);
        const float pcx = __shfl_up(scx, d, 32);
        const float pcy = __shfl_up(scy, d, 32);
        if (lane >= d) {
            scx = sa * pcx + scx;
            scy = sa * pcy + scy;
            sa  = sa * pa;
        }
    }
    if (lane == 31) { aggA[wv] = sa; aggCx[wv] = scx; aggCy[wv] = scy; }
    __syncthreads();

    if (t == 0) {   // exclusive scan over the 8 wave aggregates
        float pa = 1.0f, pcx = 0.0f, pcy = 0.0f;
        for (int v = 0; v < kWaves; ++v) {
            preA[v] = pa; preCx[v] = pcx; preCy[v] = pcy;
            const float na  = aggA[v] * pa;
            const float ncx = aggA[v] * pcx + aggCx[v];
            const float ncy = aggA[v] * pcy + aggCy[v];
            pa = na; pcx = ncx; pcy = ncy;
        }
    }
    __syncthreads();

    // exclusive within wave, then compose with wave prefix
    float ea  = __shfl_up(sa,  1, 32);
    float ecx = __shfl_up(scx, 1, 32);
    float ecy = __shfl_up(scy, 1, 32);
    if (lane == 0) { ea = 1.0f; ecx = 0.0f; ecy = 0.0f; }
    const float Pcx = ea * preCx[wv] + ecx;   // prefix value (a-part is 0 past f0)
    const float Pcy = ea * preCy[wv] + ecy;

    const float y0x = a0 * Pcx + c0x;
    const float y0y = a0 * Pcy + c0y;
    float y1x = a1s * y0x + c1x;
    float y1y = a1s * y0y + c1y;
    if (i1 == kN - 1) { y1x = ptsX[kN - 1]; y1y = ptsY[kN - 1]; }  // pin endpoint

    float4* ob4 = (float4*)(out + (size_t)bb * kN * 2);
    ob4[t] = make_float4(y0x, y0y, y1x, y1y);
}

extern "C" void kernel_launch(void* const* d_in, const int* in_sizes, int n_in,
                              void* d_out, int out_size, void* d_ws, size_t ws_size,
                              hipStream_t stream) {
    const float* wp  = (const float*)d_in[0];
    const float* sw  = (const float*)d_in[1];
    const float* W1  = (const float*)d_in[2];
    const float* B1v = (const float*)d_in[3];
    const float* W2  = (const float*)d_in[4];
    const float* B2v = (const float*)d_in[5];
    const float* W3  = (const float*)d_in[6];
    const float* B3v = (const float*)d_in[7];
    float* out = (float*)d_out;

    const int Bn = in_sizes[0] / (kN * 2);   // 4096 trajectories
    hipLaunchKernelGGL(waypoint_smoother_kernel, dim3(Bn), dim3(kThreads), 0, stream,
                       wp, sw, W1, B1v, W2, B2v, W3, B3v, out);
}
